// LSTM_57200374448605
// MI455X (gfx1250) — matrix operand order repaired and checked by reading, not compiled, not run on hardware
//
#include <hip/hip_runtime.h>

// ---- CDNA5 WMMA types -------------------------------------------------------
typedef __bf16 bf16_t;
typedef __attribute__((ext_vector_type(16))) __bf16 v16bf;
typedef __attribute__((ext_vector_type(4)))  __bf16 v4bf;
typedef __attribute__((ext_vector_type(8)))  float  v8f;
typedef __attribute__((ext_vector_type(4)))  float  vf4;

union FragBF {            // one WMMA operand: 16 bf16 per lane = 32 bytes
    v16bf v;
    vf4   f4[2];
};

// Problem sizes (fixed by the reference)
#define TT   512
#define BB   64
#define DD   512
#define HH   512
#define G4H  2048   // 4*H

__device__ __forceinline__ float sigf(float x) {
    return 1.0f / (1.0f + __expf(-x));
}

// issue one 16-byte async copy global -> LDS (ASYNCcnt-tracked, no VGPR round trip)
__device__ __forceinline__ void async_ld16(const void* gptr, void* lptr) {
    unsigned lds = (unsigned)(uintptr_t)lptr;   // LDS aperture: low 32 bits = LDS byte addr
    asm volatile("global_load_async_to_lds_b128 %0, %1, off"
                 :: "v"(lds), "v"(gptr) : "memory");
}
__device__ __forceinline__ void wait_async0() {
    asm volatile("s_wait_asynccnt 0x0" ::: "memory");
}

// ---- kernel 0: bf16 conversions + state init -------------------------------
// x[T*B*D], Wx[4H*D], Wh[4H*H] -> bf16; zero h0/c0.  One thread per vec4 unit.
__global__ __launch_bounds__(256)
void lstm_init_k(const float* __restrict__ x,  const float* __restrict__ Wx,
                 const float* __restrict__ Wh,
                 bf16_t* __restrict__ xb, bf16_t* __restrict__ Wxb,
                 bf16_t* __restrict__ Whb,
                 bf16_t* __restrict__ h0, float* __restrict__ c0) {
    int i = blockIdx.x * 256 + threadIdx.x;          // vec4 index
    if (i < (TT * BB * DD) / 4) {
        vf4 v = *(const vf4*)(x + (size_t)i * 4);
        v4bf o; o[0]=(bf16_t)v[0]; o[1]=(bf16_t)v[1]; o[2]=(bf16_t)v[2]; o[3]=(bf16_t)v[3];
        *(v4bf*)(xb + (size_t)i * 4) = o;
    }
    if (i < (G4H * DD) / 4) {
        vf4 v = *(const vf4*)(Wx + (size_t)i * 4);
        v4bf o; o[0]=(bf16_t)v[0]; o[1]=(bf16_t)v[1]; o[2]=(bf16_t)v[2]; o[3]=(bf16_t)v[3];
        *(v4bf*)(Wxb + (size_t)i * 4) = o;
    }
    if (i < (G4H * HH) / 4) {
        vf4 v = *(const vf4*)(Wh + (size_t)i * 4);
        v4bf o; o[0]=(bf16_t)v[0]; o[1]=(bf16_t)v[1]; o[2]=(bf16_t)v[2]; o[3]=(bf16_t)v[3];
        *(v4bf*)(Whb + (size_t)i * 4) = o;
    }
    if (i < BB * HH) { h0[i] = (bf16_t)0.0f; c0[i] = 0.0f; }
}

// stage one 128x32 A tile + 64x32 B tile (bf16) into LDS via async-to-LDS
__device__ __forceinline__ void stage_tile(const bf16_t* __restrict__ xb,
                                           const bf16_t* __restrict__ wxb,
                                           bf16_t (*sA)[32], bf16_t (*sB)[32],
                                           int m0, int n0, int kb, int tid) {
#pragma unroll
    for (int u = 0; u < 2; ++u) {                 // A: 512 chunks of 16B, 2/thread
        int q = tid + u * 256;
        int row = q >> 2, cc = (q & 3) * 8;
        async_ld16(xb + (size_t)(m0 + row) * DD + kb + cc, &sA[row][cc]);
    }
    {                                             // B: 256 chunks, 1/thread
        int row = tid >> 2, cc = (tid & 3) * 8;
        async_ld16(wxb + (size_t)(n0 + row) * DD + kb + cc, &sB[row][cc]);
    }
}

// ---- kernel 1: input projection xg = x @ Wx^T + b  (bf16 WMMA) -------------
// Grid (256, 32), 256 threads. WG tile: 128 rows (T*B) x 64 cols (4H), K=512.
// Double-buffered LDS; async stage of tile k+1 overlaps WMMA on tile k.
__global__ __launch_bounds__(256)
void lstm_xproj_k(const bf16_t* __restrict__ xb, const bf16_t* __restrict__ Wxb,
                  const float* __restrict__ bias, float* __restrict__ xg) {
    __shared__ bf16_t sA[2][128][32];   // 16 KB
    __shared__ bf16_t sB[2][64][32];    //  8 KB

    const int tid  = threadIdx.x;
    const int m0   = blockIdx.x * 128;
    const int n0   = blockIdx.y * 64;
    const int lane = tid & 31;
    const int wid  = tid >> 5;
    const int wm   = wid & 3;        // 4 M-subtiles of 32 rows
    const int wn   = wid >> 2;       // 2 N-subtiles of 32 cols
    const int ln   = lane & 15;
    const int kh   = lane >> 4;      // lane half selects contiguous K-half (k = 16*kh + s)

    v8f acc[2][2];
    const v8f vzero = {0.f,0.f,0.f,0.f,0.f,0.f,0.f,0.f};
    acc[0][0] = vzero; acc[0][1] = vzero; acc[1][0] = vzero; acc[1][1] = vzero;

    stage_tile(xb, Wxb, sA[0], sB[0], m0, n0, 0, tid);

    for (int kbi = 0; kbi < DD / 32; ++kbi) {
        const int cur = kbi & 1;
        wait_async0();                       // this wave's async chunks of tile kbi done
        __syncthreads();                     // every wave's chunks done; also fences
                                             // last iteration's reads of buffer 1-cur
        __builtin_amdgcn_sched_barrier(0);   // don't hoist LDS reads above wait+barrier

        if (kbi + 1 < DD / 32)               // prefetch next tile into other buffer
            stage_tile(xb, Wxb, sA[1 - cur], sB[1 - cur], m0, n0, (kbi + 1) * 32, tid);
        __builtin_amdgcn_sched_barrier(0);   // don't sink the async prefetch below WMMA

        FragBF a[2], b[2];
#pragma unroll
        for (int i = 0; i < 2; ++i) {
            int r = wm * 32 + i * 16 + ln;
            __builtin_memcpy(&a[i].f4[0], &sA[cur][r][16 * kh], 16);      // K 0..7  (+16*kh)
            __builtin_memcpy(&a[i].f4[1], &sA[cur][r][16 * kh + 8], 16); // K 8..15 (+16*kh)
        }
#pragma unroll
        for (int j = 0; j < 2; ++j) {
            int col = wn * 32 + j * 16 + ln;
            __builtin_memcpy(&b[j].f4[0], &sB[cur][col][16 * kh], 32);   // 16 contiguous K
        }
#pragma unroll
        for (int i = 0; i < 2; ++i)
#pragma unroll
            for (int j = 0; j < 2; ++j)
                acc[i][j] = __builtin_amdgcn_wmma_f32_16x16x32_bf16(
                    false, a[i].v, false, b[j].v, (short)0, acc[i][j], false, false);
    }

#pragma unroll
    for (int i = 0; i < 2; ++i)
#pragma unroll
        for (int j = 0; j < 2; ++j) {
            int col = n0 + wn * 32 + j * 16 + ln;
            float bn = bias[col];
#pragma unroll
            for (int r = 0; r < 8; ++r) {
                int row = m0 + wm * 32 + i * 16 + 8 * kh + r;  // C layout: M = r + 8*laneHalf
                xg[(size_t)row * G4H + col] = acc[i][j][r] + bn;
            }
        }
}

// ---- kernel 2: one LSTM timestep (bf16 WMMA + fused cell update) -----------
// Grid 16 WGs x 256 threads. WG owns 32 hidden units; each wave computes the
// four gate tiles (i,f,c~,o) for one (16-batch x 16-hidden) block so the cell
// update is purely lane-local. Wh slice (128 KB/WGP/step) streams from L2.
__global__ __launch_bounds__(256)
void lstm_step_k(const float* __restrict__ xg, const bf16_t* __restrict__ Whb,
                 const int* __restrict__ lengths,
                 const bf16_t* __restrict__ h_in, bf16_t* __restrict__ h_out,
                 float* __restrict__ c, float* __restrict__ out, int t) {
    const int tid   = threadIdx.x;
    const int lane  = tid & 31;
    const int wid   = tid >> 5;
    const int mt    = wid & 3;                 // batch tile (16 rows)
    const int strip = wid >> 2;                // 0..1
    const int j0    = blockIdx.x * 32 + strip * 16;
    const int ln    = lane & 15;
    const int kh    = lane >> 4;
    const int jcol  = j0 + ln;

    // seed accumulators with precomputed xg[t] (bias already folded in)
    v8f acc[4];
#pragma unroll
    for (int g = 0; g < 4; ++g)
#pragma unroll
        for (int r = 0; r < 8; ++r) {
            int m = mt * 16 + 8 * kh + r;
            acc[g][r] = xg[((size_t)t * BB + m) * G4H + g * HH + jcol];
        }

    const int arow = mt * 16 + ln;
#pragma unroll 2
    for (int kb = 0; kb < HH; kb += 32) {
        FragBF a;   // h fragment, shared across the 4 gates
        __builtin_memcpy(&a.f4[0], h_in + (size_t)arow * HH + kb + 16 * kh, 16);
        __builtin_memcpy(&a.f4[1], h_in + (size_t)arow * HH + kb + 16 * kh + 8, 16);
#pragma unroll
        for (int g = 0; g < 4; ++g) {
            FragBF b;  // Wh row (gate g, hidden jcol), 16 contiguous K bf16
            __builtin_memcpy(&b.f4[0],
                             Whb + ((size_t)(g * HH + jcol)) * HH + kb + 16 * kh, 32);
            acc[g] = __builtin_amdgcn_wmma_f32_16x16x32_bf16(
                false, a.v, false, b.v, (short)0, acc[g], false, false);
        }
    }

    // fused LSTM cell update (lane-local: same (m,j) slot in all 4 accumulators)
#pragma unroll
    for (int r = 0; r < 8; ++r) {
        int m = mt * 16 + 8 * kh + r;
        bool active = (t < lengths[m]);
        float xi = sigf(acc[0][r]);
        float xf = sigf(acc[1][r]);
        float gg = tanhf(acc[2][r]);
        float xo = sigf(acc[3][r]);
        size_t sidx = (size_t)m * HH + jcol;
        float cp = c[sidx];
        float cn = xf * cp + xi * gg;
        float ht = xo * tanhf(cn);
        size_t oidx = ((size_t)t * BB + m) * HH + jcol;
        if (active) {
            c[sidx]     = cn;
            h_out[sidx] = (bf16_t)ht;
            out[oidx]   = ht;
        } else {
            h_out[sidx] = h_in[sidx];   // carry h forward (double-buffered)
            out[oidx]   = 0.0f;         // padded output rows are zero
        }
    }
}

// ---- launcher ---------------------------------------------------------------
extern "C" void kernel_launch(void* const* d_in, const int* in_sizes, int n_in,
                              void* d_out, int out_size, void* d_ws, size_t ws_size,
                              hipStream_t stream) {
    const float* x       = (const float*)d_in[0];
    const int*   lengths = (const int*)d_in[1];
    const float* Wx      = (const float*)d_in[2];
    const float* Wh      = (const float*)d_in[3];
    const float* bias    = (const float*)d_in[4];
    float*       out     = (float*)d_out;

    char* ws = (char*)d_ws;
    size_t off = 0;
    float*  xg  = (float*)(ws + off);  off += (size_t)TT * BB * G4H * sizeof(float);  // 256 MB
    bf16_t* xb  = (bf16_t*)(ws + off); off += (size_t)TT * BB * DD * sizeof(bf16_t);  //  32 MB
    bf16_t* Wxb = (bf16_t*)(ws + off); off += (size_t)G4H * DD * sizeof(bf16_t);      //   2 MB
    bf16_t* Whb = (bf16_t*)(ws + off); off += (size_t)G4H * HH * sizeof(bf16_t);      //   2 MB
    bf16_t* h0  = (bf16_t*)(ws + off); off += (size_t)BB * HH * sizeof(bf16_t);
    bf16_t* h1  = (bf16_t*)(ws + off); off += (size_t)BB * HH * sizeof(bf16_t);
    float*  c   = (float*)(ws + off);  off += (size_t)BB * HH * sizeof(float);

    lstm_init_k<<<(TT * BB * DD / 4 + 255) / 256, 256, 0, stream>>>(
        x, Wx, Wh, xb, Wxb, Whb, h0, c);
    lstm_xproj_k<<<dim3((TT * BB) / 128, G4H / 64), 256, 0, stream>>>(xb, Wxb, bias, xg);

    for (int t = 0; t < TT; ++t) {
        bf16_t* hin  = (t & 1) ? h1 : h0;
        bf16_t* hout = (t & 1) ? h0 : h1;
        lstm_step_k<<<HH / 32, 256, 0, stream>>>(xg, Whb, lengths, hin, hout, c, out, t);
    }
}
